// BondBlock_12017318494544
// MI455X (gfx1250) — matrix operand order repaired and checked by reading, not compiled, not run on hardware
//
#include <hip/hip_runtime.h>
#include <hip/hip_bf16.h>

// ---------------------------------------------------------------------------
// CDNA5 (gfx1250) implementation of the BondBlock forward pass.
// GEMMs run on v_wmma_f32_16x16x32_bf16 (wave32 fragment layouts per ISA
// 7.12.2); scatters / softmax use float atomics; layernorm is wave-per-row.
// GEMM tile: 64(M) x 128(N), K-step 64 -> 8 WMMAs per wave per barrier pair.
// ---------------------------------------------------------------------------

typedef __attribute__((ext_vector_type(16))) __bf16 v16bf;
typedef __attribute__((ext_vector_type(8)))  float  v8f;

union BF16Frag { v16bf v; unsigned int u[8]; };

#define LN_EPS 1e-5f
enum { GM_STORE = 0, GM_ADD = 1, GM_MUL = 2 };

__device__ __forceinline__ unsigned int pack_bf16(float a0, float a1) {
    unsigned int b0 = __float_as_uint(a0);
    unsigned int b1 = __float_as_uint(a1);
    unsigned int r0 = (b0 + 0x7fffu + ((b0 >> 16) & 1u)) >> 16;   // RNE to bf16
    unsigned int r1 = (b1 + 0x7fffu + ((b1 >> 16) & 1u)) >> 16;
    return (r1 << 16) | (r0 & 0xffffu);
}

// ---------------------------------------------------------------------------
// Generic Y(M,O) = op( gather(X)(M,K) @ W(O,K)^T + bias ) via WMMA.
// Block = 256 threads = 8 waves; block tile 64(M) x 128(N); K step 64.
// Wave w: M subtile wm = w&3, N subtiles (w>>2)*4 .. +3 (4 accumulators).
// Requires K % 64 == 0, O % 128 == 0 (true here: K,O in {128,256}).
// ---------------------------------------------------------------------------
__global__ __launch_bounds__(256)
void wmma_gemm_kernel(const float* __restrict__ X, const int* __restrict__ gidx,
                      const float* __restrict__ W, const float* __restrict__ Bias,
                      float* __restrict__ Y, int M, int K, int O,
                      int mode, int relu_in)
{
    __shared__ unsigned int ldsA[64][33];    // 64 rows  x 32 bf16-pairs (K=64) + pad
    __shared__ unsigned int ldsB[128][33];   // 128 cols x 32 bf16-pairs + pad

    const int tid  = threadIdx.x;
    const int wave = tid >> 5;
    const int lane = tid & 31;
    const int bm = blockIdx.x * 64;
    const int bn = blockIdx.y * 128;

    const int wm  = wave & 3;           // M subtile 0..3
    const int wn0 = (wave >> 2) << 2;   // N subtiles {0..3} or {4..7}

    v8f c[4];
#pragma unroll
    for (int t = 0; t < 4; ++t) c[t] = (v8f){0.f,0.f,0.f,0.f,0.f,0.f,0.f,0.f};

    // A fragment (16x32 bf16): lanes 0-15 K pairs {0..3, 8..11};
    // lanes 16-31 K pairs {4..7, 12..15} (plus chunk offset).
    const int arow = (wm << 4) + (lane & 15);
    const int apb  = (lane < 16) ? 0 : 4;
    // B fragment (32x16): lane%16 = column; lanes 0-15 K pairs {0..3, 8..11}, 16-31 pairs {4..7, 12..15}.
    const int bkg  = (lane < 16) ? 0 : 4;

    for (int k0 = 0; k0 < K; k0 += 64) {
        __syncthreads();
        // ---- stage A tile: 64 rows x 32 pairs = 2048 uints (8 per thread) ----
#pragma unroll
        for (int i = 0; i < 8; ++i) {
            int idx = tid + i * 256;
            int r = idx >> 5, p = idx & 31;
            int row = bm + r; if (row >= M) row = M - 1;
            size_t src = gidx ? (size_t)gidx[row] : (size_t)row;
            const float* xp = X + src * (size_t)K + (size_t)(k0 + (p << 1));
            float a0 = xp[0], a1 = xp[1];
            if (relu_in) { a0 = fmaxf(a0, 0.f); a1 = fmaxf(a1, 0.f); }
            ldsA[r][p] = pack_bf16(a0, a1);
            if (k0 + 64 < K) __builtin_prefetch(xp + 64, 0, 0);   // next K chunk
        }
        // ---- stage B tile: 128 cols x 32 pairs = 4096 uints (16 per thread) ----
#pragma unroll
        for (int i = 0; i < 16; ++i) {
            int idx = tid + i * 256;
            int r = idx >> 5, p = idx & 31;
            const float* wp = W + (size_t)(bn + r) * (size_t)K + (size_t)(k0 + (p << 1));
            ldsB[r][p] = pack_bf16(wp[0], wp[1]);
        }
        __syncthreads();

        // ---- 2 K-chunks of 32, 4 N-subtiles each: 8 WMMAs ----
#pragma unroll
        for (int kk = 0; kk < 32; kk += 16) {
            BF16Frag af;
#pragma unroll
            for (int v = 0; v < 4; ++v) {
                af.u[v]     = ldsA[arow][kk + apb + v];
                af.u[v + 4] = ldsA[arow][kk + apb + 8 + v];
            }
#pragma unroll
            for (int t = 0; t < 4; ++t) {
                const int brow = ((wn0 + t) << 4) + (lane & 15);
                BF16Frag bf;
#pragma unroll
                for (int v = 0; v < 4; ++v) { bf.u[v] = ldsB[brow][kk + bkg + v]; bf.u[v + 4] = ldsB[brow][kk + bkg + 8 + v]; }
                c[t] = __builtin_amdgcn_wmma_f32_16x16x32_bf16(
                           false, af.v, false, bf.v, (short)0, c[t], false, false);
            }
        }
    }

    // C layout: VGPR v, lanes 0-15 -> M=v, lanes 16-31 -> M=8+v; N = lane%16.
    const int nlo   = lane & 15;
    const int rbase = bm + (wm << 4) + ((lane < 16) ? 0 : 8);
#pragma unroll
    for (int t = 0; t < 4; ++t) {
        int col = bn + ((wn0 + t) << 4) + nlo;
        float badd = Bias ? Bias[col] : 0.f;
#pragma unroll
        for (int v = 0; v < 8; ++v) {
            int row = rbase + v;
            if (row < M) {
                float* yp = Y + (size_t)row * (size_t)O + col;
                float a = c[t][v];
                if (mode == GM_STORE)    *yp = a + badd;
                else if (mode == GM_ADD) *yp += a + badd;
                else                     *yp *= a;      // elementwise gate product
            }
        }
    }
}

// ---------------------------------------------------------------------------
// Utility kernels
// ---------------------------------------------------------------------------
__global__ __launch_bounds__(256)
void zero_kernel(float* p, long n) {
    long t = (long)blockIdx.x * 256 + threadIdx.x;
    if (t < n) p[t] = 0.f;
}

__global__ __launch_bounds__(256)
void scatter_add_kernel(const float* __restrict__ src, const int* __restrict__ idx,
                        float* __restrict__ acc, int M)
{
    long t = (long)blockIdx.x * 256 + threadIdx.x;
    if (t >= (long)M * 32) return;
    int e = (int)(t >> 5);
    int j = ((int)t & 31) << 2;
    int n = idx[e];
    const float4 v = *(const float4*)(src + (size_t)e * 128 + j);
    float* d = acc + (size_t)n * 128 + j;
    atomicAdd(d + 0, v.x);
    atomicAdd(d + 1, v.y);
    atomicAdd(d + 2, v.z);
    atomicAdd(d + 3, v.w);
}

__device__ __forceinline__ float wave_sum32(float x) {
#pragma unroll
    for (int m = 16; m > 0; m >>= 1) x += __shfl_xor(x, m, 32);
    return x;
}

// h[row] += acc_l[left[row]] + acc_r[right[row]]; layernorm in place.
__global__ __launch_bounds__(256)
void combine_ln_kernel(float* __restrict__ h, const float* __restrict__ accl,
                       const float* __restrict__ accr, const int* __restrict__ left,
                       const int* __restrict__ right, const float* __restrict__ g,
                       const float* __restrict__ b, int M)
{
    int row  = blockIdx.x * 8 + (threadIdx.x >> 5);
    int lane = threadIdx.x & 31;
    if (row >= M) return;
    int j = lane << 2;
    float4 v        = *(float4*)(h + (size_t)row * 128 + j);
    const float4 al = *(const float4*)(accl + (size_t)left[row]  * 128 + j);
    const float4 ar = *(const float4*)(accr + (size_t)right[row] * 128 + j);
    v.x += al.x + ar.x; v.y += al.y + ar.y; v.z += al.z + ar.z; v.w += al.w + ar.w;
    float mu = wave_sum32(v.x + v.y + v.z + v.w) * (1.f / 128.f);
    float dx = v.x - mu, dy = v.y - mu, dz = v.z - mu, dw = v.w - mu;
    float var = wave_sum32(dx*dx + dy*dy + dz*dz + dw*dw) * (1.f / 128.f);
    float rs = rsqrtf(var + LN_EPS);
    float4 o;
    o.x = dx * rs * g[j+0] + b[j+0];
    o.y = dy * rs * g[j+1] + b[j+1];
    o.z = dz * rs * g[j+2] + b[j+2];
    o.w = dw * rs * g[j+3] + b[j+3];
    *(float4*)(h + (size_t)row * 128 + j) = o;
}

__global__ __launch_bounds__(256)
void ln_rows_kernel(const float* __restrict__ in, float* __restrict__ out,
                    const float* __restrict__ g, const float* __restrict__ b, int M)
{
    int row  = blockIdx.x * 8 + (threadIdx.x >> 5);
    int lane = threadIdx.x & 31;
    if (row >= M) return;
    int j = lane << 2;
    float4 v = *(const float4*)(in + (size_t)row * 128 + j);
    float mu = wave_sum32(v.x + v.y + v.z + v.w) * (1.f / 128.f);
    float dx = v.x - mu, dy = v.y - mu, dz = v.z - mu, dw = v.w - mu;
    float var = wave_sum32(dx*dx + dy*dy + dz*dz + dw*dw) * (1.f / 128.f);
    float rs = rsqrtf(var + LN_EPS);
    float4 o;
    o.x = dx * rs * g[j+0] + b[j+0];
    o.y = dy * rs * g[j+1] + b[j+1];
    o.z = dz * rs * g[j+2] + b[j+2];
    o.w = dw * rs * g[j+3] + b[j+3];
    *(float4*)(out + (size_t)row * 128 + j) = o;
}

// ---------------------------------------------------------------------------
// Scatter attention (H=2 heads of 64). One wave per pair.
// ---------------------------------------------------------------------------
__device__ __forceinline__ unsigned int order_key(float f) {
    unsigned int b = __float_as_uint(f);
    return (b & 0x80000000u) ? ~b : (b | 0x80000000u);
}
__device__ __forceinline__ float order_dec(unsigned int k) {
    return (k & 0x80000000u) ? __uint_as_float(k ^ 0x80000000u)
                             : __uint_as_float(~k);
}

__global__ __launch_bounds__(256)
void attn_qk_kernel(const float* __restrict__ q, const float* __restrict__ k,
                    const int* __restrict__ iq, const int* __restrict__ ik,
                    float* __restrict__ qkout, unsigned int* __restrict__ mkey, int A)
{
    int a    = blockIdx.x * 8 + (threadIdx.x >> 5);
    int lane = threadIdx.x & 31;
    if (a >= A) return;
    int eq = iq[a], ek = ik[a];
    int j = lane << 2;
    const float4 qv = *(const float4*)(q + (size_t)eq * 128 + j);
    const float4 kv = *(const float4*)(k + (size_t)ek * 128 + j);
    float d = qv.x*kv.x + qv.y*kv.y + qv.z*kv.z + qv.w*kv.w;
#pragma unroll
    for (int m = 8; m > 0; m >>= 1) d += __shfl_xor(d, m, 16);   // per-head sum
    if ((lane & 15) == 0) {
        int h = lane >> 4;                // lanes 0-15: head0, 16-31: head1
        qkout[(size_t)a * 2 + h] = d;
        atomicMax(&mkey[(size_t)eq * 2 + h], order_key(d));
    }
}

__global__ __launch_bounds__(256)
void attn_expden_kernel(float* __restrict__ qk, const int* __restrict__ iq,
                        const unsigned int* __restrict__ mkey,
                        float* __restrict__ den, int A)
{
    long t = (long)blockIdx.x * 256 + threadIdx.x;
    if (t >= (long)A * 2) return;
    int a = (int)(t >> 1), h = (int)(t & 1);
    int e = iq[a];
    float m  = order_dec(mkey[(size_t)e * 2 + h]);
    float ex = __expf(qk[t] - m);
    qk[t] = ex;
    atomicAdd(&den[(size_t)e * 2 + h], ex);
}

__global__ __launch_bounds__(256)
void attn_msg_kernel(const float* __restrict__ v, const float* __restrict__ ex,
                     const float* __restrict__ den, const int* __restrict__ iq,
                     const int* __restrict__ ik, float* __restrict__ out, int A)
{
    int a    = blockIdx.x * 8 + (threadIdx.x >> 5);
    int lane = threadIdx.x & 31;
    if (a >= A) return;
    int eq = iq[a], ek = ik[a];
    int h = lane >> 4;
    float alpha = ex[(size_t)a * 2 + h] / den[(size_t)eq * 2 + h];
    int j = lane << 2;
    const float4 vv = *(const float4*)(v + (size_t)ek * 128 + j);
    float* d = out + (size_t)eq * 128 + j;
    atomicAdd(d + 0, alpha * vv.x);
    atomicAdd(d + 1, alpha * vv.y);
    atomicAdd(d + 2, alpha * vv.z);
    atomicAdd(d + 3, alpha * vv.w);
}

// ---------------------------------------------------------------------------
// Host orchestration
// ---------------------------------------------------------------------------
extern "C" void kernel_launch(void* const* d_in, const int* in_sizes, int n_in,
                              void* d_out, int out_size, void* d_ws, size_t ws_size,
                              hipStream_t stream)
{
    const int E = in_sizes[0] / 128;
    const int N = in_sizes[2] / 256;
    const int A = in_sizes[3];

    const float* h_bond = (const float*)d_in[0];
    const int*   b_idx  = (const int*)d_in[1];
    const float* h_node = (const float*)d_in[2];
    const int* iq_l = (const int*)d_in[3];
    const int* ik_l = (const int*)d_in[4];
    const int* iq_r = (const int*)d_in[5];
    const int* ik_r = (const int*)d_in[6];
    const int* left  = b_idx;
    const int* right = b_idx + E;

    // params leaves in jax pytree (sorted-key) order, starting at d_in[7]
    const float* Wnl = (const float*)d_in[7];
    const float* Wnr = (const float*)d_in[8];
    const float* Wo  = (const float*)d_in[9];
    const float* Ws  = (const float*)d_in[10];
    const float* bnl = (const float*)d_in[11];
    const float* bnr = (const float*)d_in[12];
    const float* bo  = (const float*)d_in[13];
    const float* bs  = (const float*)d_in[14];
    // ffn_l: W1, W2, Wb, Wn, b1, b2
    const float* fl_W1 = (const float*)d_in[15];
    const float* fl_W2 = (const float*)d_in[16];
    const float* fl_Wb = (const float*)d_in[17];
    const float* fl_Wn = (const float*)d_in[18];
    const float* fl_b1 = (const float*)d_in[19];
    const float* fl_b2 = (const float*)d_in[20];
    const float* fr_W1 = (const float*)d_in[21];
    const float* fr_W2 = (const float*)d_in[22];
    const float* fr_Wb = (const float*)d_in[23];
    const float* fr_Wn = (const float*)d_in[24];
    const float* fr_b1 = (const float*)d_in[25];
    const float* fr_b2 = (const float*)d_in[26];
    const float* ln1_b = (const float*)d_in[27];
    const float* ln1_g = (const float*)d_in[28];
    const float* ln2_b = (const float*)d_in[29];
    const float* ln2_g = (const float*)d_in[30];
    const float* ln_b  = (const float*)d_in[31];
    const float* ln_g  = (const float*)d_in[32];
    // qkv: Wk, Wq, Wv, bk, bq, bv
    const float* ql_Wk = (const float*)d_in[33];
    const float* ql_Wq = (const float*)d_in[34];
    const float* ql_Wv = (const float*)d_in[35];
    const float* ql_bk = (const float*)d_in[36];
    const float* ql_bq = (const float*)d_in[37];
    const float* ql_bv = (const float*)d_in[38];
    const float* qr_Wk = (const float*)d_in[39];
    const float* qr_Wq = (const float*)d_in[40];
    const float* qr_Wv = (const float*)d_in[41];
    const float* qr_bk = (const float*)d_in[42];
    const float* qr_bq = (const float*)d_in[43];
    const float* qr_bv = (const float*)d_in[44];

    // ---- workspace carve-up (floats) ----
    float* ws      = (float*)d_ws;
    float* P_inter = ws;                              // E*256
    float* P_h1    = P_inter + (size_t)E * 256;       // E*256
    float* P_m     = P_h1    + (size_t)E * 256;       // E*128 (reused by attention)
    float* P_h     = P_m     + (size_t)E * 128;       // E*128
    float* acc_l   = P_h     + (size_t)E * 128;       // N*128
    float* acc_r   = acc_l   + (size_t)N * 128;       // N*128
    // attention aliases (buffers free at that point)
    float* P_q   = P_inter;
    float* P_k   = P_inter + (size_t)E * 128;
    float* P_v   = P_h1;
    float* P_att = P_h1 + (size_t)E * 128;
    float* P_ex  = P_m;                               // A*2
    float* P_mm  = P_m + (size_t)A * 2;               // E*2 (uint keys)
    float* P_den = P_mm + (size_t)E * 2;              // E*2

    auto gemm = [&](const float* X, const int* g, const float* W, const float* bias,
                    float* Y, int M, int K, int O, int mode, int relu) {
        dim3 gr((unsigned)((M + 63) / 64), (unsigned)(O / 128));
        wmma_gemm_kernel<<<gr, 256, 0, stream>>>(X, g, W, bias, Y, M, K, O, mode, relu);
    };
    auto zero = [&](float* p, long n) {
        zero_kernel<<<(unsigned)((n + 255) / 256), 256, 0, stream>>>(p, n);
    };

    zero(acc_l, (long)N * 128);
    zero(acc_r, (long)N * 128);

    // ---- BondFFN left: inter = (h_bond@Wb^T) * (node[left]@Wn^T); MLP; scatter to right
    gemm(h_bond, nullptr, fl_Wb, nullptr, P_inter, E, 128, 256, GM_STORE, 0);
    gemm(h_node, left,    fl_Wn, nullptr, P_inter, E, 256, 256, GM_MUL,   0);
    gemm(P_inter, nullptr, fl_W1, fl_b1,  P_h1,    E, 256, 256, GM_STORE, 0);
    gemm(P_h1,   nullptr, fl_W2, fl_b2,   P_m,     E, 256, 128, GM_STORE, 1); // relu on input
    scatter_add_kernel<<<(unsigned)(((long)E * 32 + 255) / 256), 256, 0, stream>>>(P_m, right, acc_l, E);

    // ---- BondFFN right
    gemm(h_bond, nullptr, fr_Wb, nullptr, P_inter, E, 128, 256, GM_STORE, 0);
    gemm(h_node, right,   fr_Wn, nullptr, P_inter, E, 256, 256, GM_MUL,   0);
    gemm(P_inter, nullptr, fr_W1, fr_b1,  P_h1,    E, 256, 256, GM_STORE, 0);
    gemm(P_h1,   nullptr, fr_W2, fr_b2,   P_m,     E, 256, 128, GM_STORE, 1);
    scatter_add_kernel<<<(unsigned)(((long)E * 32 + 255) / 256), 256, 0, stream>>>(P_m, left, acc_r, E);

    // ---- h = m_l[left] + m_r[right] + node[left]@Wnl^T+bnl + node[right]@Wnr^T+bnr + h_bond@Ws^T+bs
    gemm(h_node, left,    Wnl, bnl, P_h, E, 256, 128, GM_STORE, 0);
    gemm(h_node, right,   Wnr, bnr, P_h, E, 256, 128, GM_ADD,   0);
    gemm(h_bond, nullptr, Ws,  bs,  P_h, E, 128, 128, GM_ADD,   0);
    combine_ln_kernel<<<(unsigned)((E + 7) / 8), 256, 0, stream>>>(
        P_h, acc_l, acc_r, left, right, ln_g, ln_b, E);

    // ---- two scatter-attention passes
    auto attn_pass = [&](const int* iq, const int* ik,
                         const float* Wk, const float* Wq, const float* Wv,
                         const float* bk, const float* bq, const float* bv,
                         const float* g, const float* b) {
        gemm(P_h, nullptr, Wq, bq, P_q, E, 128, 128, GM_STORE, 0);
        gemm(P_h, nullptr, Wk, bk, P_k, E, 128, 128, GM_STORE, 0);
        gemm(P_h, nullptr, Wv, bv, P_v, E, 128, 128, GM_STORE, 0);
        zero(P_mm, (long)E * 2);
        zero(P_den, (long)E * 2);
        zero(P_att, (long)E * 128);
        attn_qk_kernel<<<(unsigned)((A + 7) / 8), 256, 0, stream>>>(
            P_q, P_k, iq, ik, P_ex, (unsigned int*)P_mm, A);
        attn_expden_kernel<<<(unsigned)(((long)A * 2 + 255) / 256), 256, 0, stream>>>(
            P_ex, iq, (const unsigned int*)P_mm, P_den, A);
        attn_msg_kernel<<<(unsigned)((A + 7) / 8), 256, 0, stream>>>(
            P_v, P_ex, P_den, iq, ik, P_att, A);
        ln_rows_kernel<<<(unsigned)((E + 7) / 8), 256, 0, stream>>>(P_att, P_h, g, b, E);
    };
    attn_pass(iq_l, ik_l, ql_Wk, ql_Wq, ql_Wv, ql_bk, ql_bq, ql_bv, ln1_g, ln1_b);
    attn_pass(iq_r, ik_r, qr_Wk, qr_Wq, qr_Wv, qr_bk, qr_bq, qr_bv, ln2_g, ln2_b);

    // ---- out = relu(h) @ Wo^T + bo
    gemm(P_h, nullptr, Wo, bo, (float*)d_out, E, 128, 128, GM_STORE, 1);

    (void)n_in; (void)out_size; (void)ws_size;
}
